// LBM_NMAR_17068200034388
// MI455X (gfx1250) — matrix operand order repaired and checked by reading, not compiled, not run on hardware
//
#include <hip/hip_runtime.h>

// ---------------- problem constants ----------------
#define N1c 800
#define N2c 800
#define NQc 6
#define NLc 6
#define NPc 96000
#define NNc 96000
#define NZc 448000
#define LOG2PIf 1.8378770664093453f

// var_params layout offsets
#define OFF_NU_A  0
#define OFF_LR_A  (N1c)
#define OFF_NU_B  (2*N1c)
#define OFF_LR_B  (3*N1c)
#define OFF_NU_P  (4*N1c)
#define OFF_LR_P  (4*N1c + N2c)
#define OFF_NU_Q  (4*N1c + 2*N2c)
#define OFF_LR_Q  (4*N1c + 3*N2c)
#define OFF_T1    (4*N1c + 4*N2c)
#define OFF_T2    (OFF_T1 + N1c*(NQc-1))

// ---------------- workspace layout (floats) ----------------
#define WS_RHO_A 0
#define WS_RHO_B (WS_RHO_A + N1c)
#define WS_RHO_P (WS_RHO_B + N1c)
#define WS_RHO_Q (WS_RHO_P + N2c)
#define WS_TAU1  (WS_RHO_Q + N2c)
#define WS_TAU2  (WS_TAU1 + N1c*6)
#define WS_SP    (WS_TAU2 + N2c*6)   // 16x16 f32 WMMA accumulator (P edges)
#define WS_SM    (WS_SP + 256)       // 16x16 f32 WMMA accumulator (M edges)
#define WS_ACC   (WS_SM + 256)
// scalar accumulator slots within WS_ACC
#define A_SCALP 0
#define A_SCALM 1
#define A_TERMZ 2
#define A_ENT1  3
#define A_ENT2  4
#define A_SUMA  5
#define A_SUMB  6
#define A_SUMP  7
#define A_SUMQ  8
#define A_LRA   9
#define A_LRB   10
#define A_LRP   11
#define A_LRQ   12
#define A_T1SUM 13   // 6 slots
#define A_T2SUM 19   // 6 slots
#define N_ACC   25
#define WS_TOTAL (WS_ACC + N_ACC)

typedef __attribute__((ext_vector_type(16))) _Float16 v16h;
typedef __attribute__((ext_vector_type(8)))  _Float16 v8h;
typedef __attribute__((ext_vector_type(8)))  float    v8f;

__device__ __forceinline__ float sigmoidf_(float x) { return 1.f / (1.f + expf(-x)); }
// numerically stable log(sigmoid(x))
__device__ __forceinline__ float log_sigmoidf_(float x) {
    return (x >= 0.f) ? -log1pf(expf(-x)) : (x - log1pf(expf(x)));
}
__device__ __forceinline__ float wave_sum(float v) {
#pragma unroll
    for (int m = 16; m >= 1; m >>= 1) v += __shfl_xor(v, m, 32);
    return v;
}

// ---------------- zero accumulators (ws is poisoned by harness) ----------------
__global__ void k_init(float* __restrict__ ws) {
    for (int t = threadIdx.x; t < (WS_TOTAL - WS_SP); t += blockDim.x)
        ws[WS_SP + t] = 0.f;
}

// ---------------- node-level precompute + small reductions ----------------
__global__ void k_nodes(const float* __restrict__ vp, float* __restrict__ ws) {
    int t = blockIdx.x * blockDim.x + threadIdx.x;
    if (t >= N1c) return;  // N1 == N2

    // rho = exp(log-rho), plus gauss_ell / entropy reductions
    float la = vp[OFF_LR_A + t], lb = vp[OFF_LR_B + t];
    float lp = vp[OFF_LR_P + t], lq = vp[OFF_LR_Q + t];
    float ra = expf(la), rb = expf(lb), rp = expf(lp), rq = expf(lq);
    ws[WS_RHO_A + t] = ra; ws[WS_RHO_B + t] = rb;
    ws[WS_RHO_P + t] = rp; ws[WS_RHO_Q + t] = rq;
    float na = vp[OFF_NU_A + t], nb = vp[OFF_NU_B + t];
    float np_ = vp[OFF_NU_P + t], nq_ = vp[OFF_NU_Q + t];
    atomicAdd(&ws[WS_ACC + A_SUMA], ra + na * na);
    atomicAdd(&ws[WS_ACC + A_SUMB], rb + nb * nb);
    atomicAdd(&ws[WS_ACC + A_SUMP], rp + np_ * np_);
    atomicAdd(&ws[WS_ACC + A_SUMQ], rq + nq_ * nq_);
    atomicAdd(&ws[WS_ACC + A_LRA], la);
    atomicAdd(&ws[WS_ACC + A_LRB], lb);
    atomicAdd(&ws[WS_ACC + A_LRP], lp);
    atomicAdd(&ws[WS_ACC + A_LRQ], lq);

    // tau_1 softmax row (5 logits + implicit 0)
    {
        float lg[6]; lg[5] = 0.f;
#pragma unroll
        for (int k = 0; k < 5; ++k) lg[k] = vp[OFF_T1 + t * 5 + k];
        float mx = lg[5];
#pragma unroll
        for (int k = 0; k < 5; ++k) mx = fmaxf(mx, lg[k]);
        float ex[6], s = 0.f;
#pragma unroll
        for (int k = 0; k < 6; ++k) { ex[k] = expf(lg[k] - mx); s += ex[k]; }
        float inv = 1.f / s, ent = 0.f;
#pragma unroll
        for (int k = 0; k < 6; ++k) {
            float tau = ex[k] * inv;
            ws[WS_TAU1 + t * 6 + k] = tau;
            ent += tau * logf(tau);
            atomicAdd(&ws[WS_ACC + A_T1SUM + k], tau);
        }
        atomicAdd(&ws[WS_ACC + A_ENT1], ent);
    }
    // tau_2 softmax row
    {
        float lg[6]; lg[5] = 0.f;
#pragma unroll
        for (int k = 0; k < 5; ++k) lg[k] = vp[OFF_T2 + t * 5 + k];
        float mx = lg[5];
#pragma unroll
        for (int k = 0; k < 5; ++k) mx = fmaxf(mx, lg[k]);
        float ex[6], s = 0.f;
#pragma unroll
        for (int k = 0; k < 6; ++k) { ex[k] = expf(lg[k] - mx); s += ex[k]; }
        float inv = 1.f / s, ent = 0.f;
#pragma unroll
        for (int k = 0; k < 6; ++k) {
            float tau = ex[k] * inv;
            ws[WS_TAU2 + t * 6 + k] = tau;
            ent += tau * logf(tau);
            atomicAdd(&ws[WS_ACC + A_T2SUM + k], tau);
        }
        atomicAdd(&ws[WS_ACC + A_ENT2], ent);
    }
}

// ---------------- P/M edge kernel: WMMA outer-product GEMM + scal sum ---------
// S = sum_e tau1[i_e] (x) tau2[j_e], computed as A(16x32 f16) x B(32x16 f16)
// per 32-edge chunk with V_WMMA_F32_16X16X32_F16; rows/cols >= 6 padded zero.
// Both staging tiles are stored 16(row) x 32(k) so every lane's fragment is
// two contiguous, 16B-aligned ds_load_b128 reads.
__global__ void __launch_bounds__(256) k_pm(
    const float* __restrict__ vp, const float* __restrict__ mp,
    const int* __restrict__ ii, const int* __restrict__ jj, int nE, float sgn,
    float* __restrict__ ws, int sOff, int scalSlot) {
    __shared__ __attribute__((aligned(32))) _Float16 ldsA[8][512];  // [q][k]  = A row-major
    __shared__ __attribute__((aligned(32))) _Float16 ldsB[8][512];  // [l][k]  = B col-major
    const int lane = threadIdx.x & 31;
    const int wv = threadIdx.x >> 5;
    const int gw = blockIdx.x * (blockDim.x >> 5) + wv;
    const int nw = gridDim.x * (blockDim.x >> 5);

    // zero this wave's staging tiles (pad rows/cols stay zero forever)
    for (int t = lane; t < 512; t += 32) {
        ldsA[wv][t] = (_Float16)0.f;
        ldsB[wv][t] = (_Float16)0.f;
    }

    const float mu = mp[0];
    const float* nu_a = vp + OFF_NU_A;
    const float* nu_b = vp + OFF_NU_B;
    const float* nu_p = vp + OFF_NU_P;
    const float* nu_q = vp + OFF_NU_Q;
    const float* rho_a = ws + WS_RHO_A;
    const float* rho_b = ws + WS_RHO_B;
    const float* rho_p = ws + WS_RHO_P;
    const float* rho_q = ws + WS_RHO_Q;

    v8f c = {};
    float sacc = 0.f;
    const int nchunk = (nE + 31) >> 5;
    for (int ch = gw; ch < nchunk; ch += nw) {
        int e = (ch << 5) + lane;
        _Float16 t1h[6], t2h[6];
        float scal = 0.f;
        if (e < nE) {
            int i = ii[e], j = jj[e];
#pragma unroll
            for (int q = 0; q < 6; ++q) t1h[q] = (_Float16)ws[WS_TAU1 + i * 6 + q];
#pragma unroll
            for (int l = 0; l < 6; ++l) t2h[l] = (_Float16)ws[WS_TAU2 + j * 6 + l];
            float z = mu + nu_a[i] + sgn * nu_b[i] + nu_p[j] + sgn * nu_q[j];
            float s = sigmoidf_(z);
            float sv = rho_a[i] + rho_p[j] + rho_b[i] + rho_q[j];
            scal = log_sigmoidf_(z) - 0.5f * s * (1.f - s) * sv;
        } else {
#pragma unroll
            for (int q = 0; q < 6; ++q) { t1h[q] = (_Float16)0.f; t2h[q] = (_Float16)0.f; }
        }
        sacc += scal;
        // stage: A[q][k=lane] = tau1_q , B(colmajor)[l][k=lane] = tau2_l
#pragma unroll
        for (int q = 0; q < 6; ++q) ldsA[wv][q * 32 + lane] = t1h[q];
#pragma unroll
        for (int l = 0; l < 6; ++l) ldsB[wv][l * 32 + lane] = t2h[l];
        // A fragment (ISA 7.12.2, 16-bit A 16x32): row = lane&15;
        // elems 0..7 = K kb..kb+7, elems 8..15 = K 16+kb..16+kb+7 (kb = 0|8)
        int row = lane & 15;
        int kb = (lane < 16) ? 0 : 8;
        v8h alo = *(const v8h*)&ldsA[wv][row * 32 + kb];
        v8h ahi = *(const v8h*)&ldsA[wv][row * 32 + 16 + kb];
        v16h a = __builtin_shufflevector(alo, ahi,
                                         0, 1, 2, 3, 4, 5, 6, 7,
                                         8, 9, 10, 11, 12, 13, 14, 15);
        // B fragment: col = lane&15; elems 0..7 = K kb2..kb2+7,
        // elems 8..15 = K 16+kb2..16+kb2+7 (kb2 = 0|8), same order as A
        int kb2 = (lane < 16) ? 0 : 8;
        v8h blo = *(const v8h*)&ldsB[wv][row * 32 + kb2];
        v8h bhi = *(const v8h*)&ldsB[wv][row * 32 + 16 + kb2];
        v16h b = __builtin_shufflevector(blo, bhi,
                                         0, 1, 2, 3, 4, 5, 6, 7,
                                         8, 9, 10, 11, 12, 13, 14, 15);
        c = __builtin_amdgcn_wmma_f32_16x16x32_f16(
            false, a, false, b, (short)0, c, false, false);
    }
    sacc = wave_sum(sacc);
    if (lane == 0) atomicAdd(&ws[WS_ACC + scalSlot], sacc);
    // C/D layout: VGPR v holds M = v + (lane<16?0:8), N = lane&15
    int m0 = (lane < 16) ? 0 : 8;
    int n = lane & 15;
#pragma unroll
    for (int v = 0; v < 8; ++v)
        atomicAdd(&ws[sOff + (m0 + v) * 16 + n], c[v]);
}

// ---------------- Z edge kernel: per-edge 6x6 transcendental contraction -----
__global__ void __launch_bounds__(256) k_z(
    const float* __restrict__ vp, const float* __restrict__ mp,
    const int* __restrict__ ii, const int* __restrict__ jj,
    float* __restrict__ ws) {
    __shared__ float pi_s[36];
    if (threadIdx.x < 36) pi_s[threadIdx.x] = sigmoidf_(mp[15 + threadIdx.x]);
    __syncthreads();

    const float mu = mp[0];
    const float* nu_a = vp + OFF_NU_A;
    const float* nu_b = vp + OFF_NU_B;
    const float* nu_p = vp + OFF_NU_P;
    const float* nu_q = vp + OFF_NU_Q;
    const float* rho_a = ws + WS_RHO_A;
    const float* rho_b = ws + WS_RHO_B;
    const float* rho_p = ws + WS_RHO_P;
    const float* rho_q = ws + WS_RHO_Q;

    float acc = 0.f;
    int stride = gridDim.x * blockDim.x;
    for (int e = blockIdx.x * blockDim.x + threadIdx.x; e < NZc; e += stride) {
        int i = ii[e], j = jj[e];
        float xz = nu_a[i] + nu_p[j];
        float yz = nu_b[i] + nu_q[j];
        float s1 = sigmoidf_(mu + xz + yz);
        float s2 = sigmoidf_(mu + xz - yz);
        float d1 = s1 * (1.f - s1), d2 = s2 * (1.f - s2);
        float dd1 = d1 * (1.f - 2.f * s1), dd2 = d2 * (1.f - 2.f * s2);
        float va = rho_a[i] + rho_p[j];
        float vb = rho_b[i] + rho_q[j];
        float t1[6], t2[6];
#pragma unroll
        for (int q = 0; q < 6; ++q) t1[q] = ws[WS_TAU1 + i * 6 + q];
#pragma unroll
        for (int l = 0; l < 6; ++l) t2[l] = ws[WS_TAU2 + j * 6 + l];
        float ae = 0.f;
#pragma unroll
        for (int q = 0; q < 6; ++q) {
            float tq = t1[q];
#pragma unroll
            for (int l = 0; l < 6; ++l) {
                float p = pi_s[q * 6 + l], om = 1.f - p;
                float g = 1.f - p * s1 - om * s2;
                float gpp = -p * dd1 - om * dd2;
                float pd = p * d1, od = om * d2;
                float gx = -(pd + od);
                float gy = od - pd;
                float ig = 1.f / g;
                float dx = (gpp - gx * gx * ig) * ig;  // (gpp*g - gx^2)/g^2
                float dy = (gpp - gy * gy * ig) * ig;
                float M = logf(g) + 0.5f * (dx * va + dy * vb);
                ae += tq * t2[l] * M;
            }
        }
        acc += ae;
    }
    acc = wave_sum(acc);
    if ((threadIdx.x & 31) == 0) atomicAdd(&ws[WS_ACC + A_TERMZ], acc);
}

// ---------------- final scalar assembly ----------------
__global__ void k_final(const float* __restrict__ vp, const float* __restrict__ mp,
                        const float* __restrict__ ws, float* __restrict__ out) {
    if (threadIdx.x != 0 || blockIdx.x != 0) return;
    const float* acc = ws + WS_ACC;

    float entropy = 0.5f * (2 * N1c + 2 * N2c) * (LOG2PIf + 1.f)
                  + 0.5f * (acc[A_LRA] + acc[A_LRB] + acc[A_LRP] + acc[A_LRQ])
                  - acc[A_ENT1] - acc[A_ENT2];

    float lsa = mp[1], lsb = mp[2], lsp = mp[3], lsq = mp[4];
    float ell_a = -0.5f * N1c * (LOG2PIf + lsa) - acc[A_SUMA] / (2.f * expf(lsa));
    float ell_b = -0.5f * N1c * (LOG2PIf + lsb) - acc[A_SUMB] / (2.f * expf(lsb));
    float ell_p = -0.5f * N2c * (LOG2PIf + lsp) - acc[A_SUMP] / (2.f * expf(lsp));
    float ell_q = -0.5f * N2c * (LOG2PIf + lsq) - acc[A_SUMQ] / (2.f * expf(lsq));

    // log-softmax alphas (5 logits + implicit 0)
    float ell_y1 = 0.f, ell_y2 = 0.f;
    {
        float x[6]; x[5] = 0.f;
        for (int k = 0; k < 5; ++k) x[k] = mp[5 + k];
        float mx = x[5];
        for (int k = 0; k < 5; ++k) mx = fmaxf(mx, x[k]);
        float s = 0.f;
        for (int k = 0; k < 6; ++k) s += expf(x[k] - mx);
        float lse = mx + logf(s);
        for (int k = 0; k < 6; ++k) ell_y1 += acc[A_T1SUM + k] * (x[k] - lse);
    }
    {
        float x[6]; x[5] = 0.f;
        for (int k = 0; k < 5; ++k) x[k] = mp[10 + k];
        float mx = x[5];
        for (int k = 0; k < 5; ++k) mx = fmaxf(mx, x[k]);
        float s = 0.f;
        for (int k = 0; k < 6; ++k) s += expf(x[k] - mx);
        float lse = mx + logf(s);
        for (int k = 0; k < 6; ++k) ell_y2 += acc[A_T2SUM + k] * (x[k] - lse);
    }

    float tp = acc[A_SCALP], tm = acc[A_SCALM];
    for (int q = 0; q < 6; ++q)
        for (int l = 0; l < 6; ++l) {
            float z = mp[15 + q * 6 + l];
            tp += ws[WS_SP + q * 16 + l] * log_sigmoidf_(z);    // log(pi)
            tm += ws[WS_SM + q * 16 + l] * log_sigmoidf_(-z);   // log1p(-pi)
        }
    float exp_x = tp + tm + acc[A_TERMZ];
    if (!(exp_x < 0.f)) exp_x = __builtin_inff();

    float res = entropy + ell_a + ell_b + ell_p + ell_q + ell_y1 + ell_y2 + exp_x;
    out[0] = -res;
}

// ---------------- launch ----------------
extern "C" void kernel_launch(void* const* d_in, const int* in_sizes, int n_in,
                              void* d_out, int out_size, void* d_ws, size_t ws_size,
                              hipStream_t stream) {
    const float* vp = (const float*)d_in[0];
    const float* mp = (const float*)d_in[1];
    const int* ip_i = (const int*)d_in[2];
    const int* ip_j = (const int*)d_in[3];
    const int* in_i = (const int*)d_in[4];
    const int* in_j = (const int*)d_in[5];
    const int* iz_i = (const int*)d_in[6];
    const int* iz_j = (const int*)d_in[7];
    float* ws = (float*)d_ws;
    float* out = (float*)d_out;

    k_init<<<1, 256, 0, stream>>>(ws);
    k_nodes<<<(N1c + 255) / 256, 256, 0, stream>>>(vp, ws);
    k_pm<<<48, 256, 0, stream>>>(vp, mp, ip_i, ip_j, NPc, 1.f, ws, WS_SP, A_SCALP);
    k_pm<<<48, 256, 0, stream>>>(vp, mp, in_i, in_j, NNc, -1.f, ws, WS_SM, A_SCALM);
    k_z<<<896, 256, 0, stream>>>(vp, mp, iz_i, iz_j, ws);
    k_final<<<1, 32, 0, stream>>>(vp, mp, ws, out);
}
